// MambaBlock_84000970375318
// MI455X (gfx1250) — compile-verified, matrix operand order checked by reading, not run on hardware
//
#include <hip/hip_runtime.h>
#include <hip/hip_bf16.h>
#include <stdint.h>

// ---------------------------------------------------------------------------
// Mamba block for MI455X (gfx1250, wave32).
//   * GEMMs: V_WMMA_F32_16X16X32_BF16, LDS-tiled (64x128 block tile, 8 waves,
//     2x2 register blocking per wave), global->LDS via GLOBAL_LOAD_ASYNC_TO_LDS
//     (ASYNCcnt + s_wait_asynccnt) when the toolchain exposes the builtin.
//   * Selective scan: 16 fp32 states per thread in registers, B_t/C_t staged
//     in LDS 256 timesteps at a time (barrier-free inner recurrence).
// ---------------------------------------------------------------------------

#define DIM      1024
#define D_STATE  16
#define D_CONV   4
#define D_INNER  2048
#define DT_RANK  64
#define B_SZ     2
#define L_SEQ    1024
#define NTOK     (B_SZ * L_SEQ)     // 2048 tokens
#define EPS      1e-5f

typedef __attribute__((ext_vector_type(16))) __bf16 v16bf;
typedef __attribute__((ext_vector_type(8)))  float  v8f;
typedef __attribute__((ext_vector_type(4)))  int    v4i;

#if __has_builtin(__builtin_amdgcn_global_load_async_to_lds_b128) && \
    __has_builtin(__builtin_amdgcn_s_wait_asynccnt)
#define USE_ASYNC_LDS 1
typedef __attribute__((address_space(1))) v4i* g_v4i_ptr;
typedef __attribute__((address_space(3))) v4i* l_v4i_ptr;
#endif

// ---------------- helpers ----------------

__device__ __forceinline__ uint16_t f2bf(float f) {
  uint32_t u = __float_as_uint(f);
  u += 0x7FFFu + ((u >> 16) & 1u);    // round-to-nearest-even
  return (uint16_t)(u >> 16);
}

__device__ __forceinline__ float sigmoidf_(float x) {
  return 1.0f / (1.0f + __expf(-x));
}

// 16-byte global -> LDS copy (async-to-LDS on CDNA5, else via VGPR)
__device__ __forceinline__ void cp16_g2l(const uint16_t* g, uint16_t* l) {
#ifdef USE_ASYNC_LDS
  __builtin_amdgcn_global_load_async_to_lds_b128(
      (g_v4i_ptr)(uintptr_t)g,
      (l_v4i_ptr)(uintptr_t)(uint32_t)(uintptr_t)l,
      0, 0);
#else
  *(uint4*)l = *(const uint4*)g;
#endif
}

__device__ __forceinline__ void cp_wait() {
#ifdef USE_ASYNC_LDS
  __builtin_amdgcn_s_wait_asynccnt(0);
#endif
}

// ---------------- elementwise kernels ----------------

__global__ void k_cvt_bf16(const float* __restrict__ src, uint16_t* __restrict__ dst, int n) {
  int i = blockIdx.x * blockDim.x + threadIdx.x;
  if (i < n) dst[i] = f2bf(src[i]);
}

// RMSNorm: one block per token (DIM=1024, 256 thr, 4 elems/thr) -> bf16
__global__ void k_rmsnorm(const float* __restrict__ x, const float* __restrict__ w,
                          uint16_t* __restrict__ xn_bf) {
  __shared__ float red[256];
  const int row = blockIdx.x;
  const int tid = threadIdx.x;
  const float* xr = x + (size_t)row * DIM;
  float ss = 0.0f;
#pragma unroll
  for (int j = 0; j < 4; ++j) {
    float v = xr[tid + j * 256];
    ss += v * v;
  }
  red[tid] = ss;
  __syncthreads();
  for (int s = 128; s > 0; s >>= 1) {
    if (tid < s) red[tid] += red[tid + s];
    __syncthreads();
  }
  const float scale = rsqrtf(red[0] * (1.0f / DIM) + EPS);
  uint16_t* orow = xn_bf + (size_t)row * DIM;
#pragma unroll
  for (int j = 0; j < 4; ++j) {
    int idx = tid + j * 256;
    orow[idx] = f2bf(xr[idx] * scale * w[idx]);
  }
}

// depthwise causal conv over time + SiLU.  xi = first D_INNER cols of xz.
__global__ void k_conv_silu(const float* __restrict__ xz, const float* __restrict__ cw,
                            const float* __restrict__ cb, float* __restrict__ u,
                            uint16_t* __restrict__ u_bf) {
  int i = blockIdx.x * blockDim.x + threadIdx.x;          // [b][t][d]
  if (i >= B_SZ * L_SEQ * D_INNER) return;
  const int d = i % D_INNER;
  const int t = (i / D_INNER) % L_SEQ;
  const int b = i / (D_INNER * L_SEQ);
  float acc = cb[d];
#pragma unroll
  for (int j = 0; j < D_CONV; ++j) {
    int ts = t - (D_CONV - 1) + j;
    if (ts >= 0)
      acc += cw[d * D_CONV + j] *
             xz[((size_t)(b * L_SEQ + ts)) * (2 * D_INNER) + d];
  }
  float s = acc * sigmoidf_(acc);
  u[i] = s;
  u_bf[i] = f2bf(s);
}

// slice dt (cols 0..63 of x_dbl, row stride 96) -> bf16 compact [NTOK,64]
__global__ void k_dt_slice(const float* __restrict__ xdbl, uint16_t* __restrict__ dt_bf) {
  int i = blockIdx.x * blockDim.x + threadIdx.x;
  if (i >= NTOK * DT_RANK) return;
  int m = i / DT_RANK, r = i % DT_RANK;
  dt_bf[i] = f2bf(xdbl[(size_t)m * (DT_RANK + 2 * D_STATE) + r]);
}

// delta = softplus(raw + bias[d]) in place
__global__ void k_softplus_bias(float* __restrict__ delta, const float* __restrict__ bias) {
  int i = blockIdx.x * blockDim.x + threadIdx.x;
  if (i >= NTOK * D_INNER) return;
  float v = delta[i] + bias[i % D_INNER];
  delta[i] = (v > 20.0f) ? v : log1pf(__expf(v));
}

// y = (ys + u*D) * silu(z) -> bf16 for out_proj
__global__ void k_gate(const float* __restrict__ ys, const float* __restrict__ u,
                       const float* __restrict__ Dp, const float* __restrict__ xz,
                       uint16_t* __restrict__ y_bf) {
  int i = blockIdx.x * blockDim.x + threadIdx.x;
  if (i >= NTOK * D_INNER) return;
  int m = i / D_INNER, d = i % D_INNER;
  float z = xz[(size_t)m * (2 * D_INNER) + D_INNER + d];
  float y = ys[i] + u[i] * Dp[d];
  y_bf[i] = f2bf(y * z * sigmoidf_(z));
}

// out = x + proj (residual)
__global__ void k_residual(const float* __restrict__ x, const float* __restrict__ proj,
                           float* __restrict__ out) {
  int i = blockIdx.x * blockDim.x + threadIdx.x;
  if (i < NTOK * DIM) out[i] = x[i] + proj[i];
}

// ---------------------------------------------------------------------------
// LDS fragment layouts for V_WMMA_F32_16X16X32_BF16 (wave32):
//  A (16x32, 16-bit): lane -> row m = lane%16; VGPR v holds K-pair at
//    k = (v/4)*16 + (lane/16)*8 + (v%4)*2  => two contiguous 16B runs per
//    lane at k = half*8 and k = 16 + half*8.
//  B mirrors A with lane -> column n.  D: 8 VGPRs, row = half*8+r, col=lane%16.
// ---------------------------------------------------------------------------

union Frag { uint4 q[2]; v16bf h; };

__device__ __forceinline__ Frag lds_frag(const uint16_t* base, int stride,
                                         int row, int half) {
  Frag f;
  const uint16_t* p = base + row * stride + half * 8;
  f.q[0] = *(const uint4*)p;          // K group 0..15  (this half's 8)
  f.q[1] = *(const uint4*)(p + 16);   // K group 16..31 (this half's 8)
  return f;
}

// ---------------- tiled WMMA GEMM ----------------
// C[M,N] = A[M,K] * W[N,K]^T, bf16 in / f32 out.
// Block: 256 threads = 8 waves (2 M x 4 N). Block tile 64(M) x 128(N).
// Each wave: 32x32 = 2x2 WMMA tiles.  K staged in LDS 32 at a time.
// Requires M%64==0, N%128==0, K%32==0.
#define BM 64
#define BN 128
#define KC 32
#define LDT (KC + 8)      // LDS row stride in bf16 elems (pad vs bank conflicts)

__global__ void __launch_bounds__(256)
k_gemm_tiled(const uint16_t* __restrict__ A, const uint16_t* __restrict__ W,
             float* __restrict__ C, int M, int N, int K) {
  __shared__ uint16_t sA[BM * LDT];   // 64 x 32 (+pad)
  __shared__ uint16_t sW[BN * LDT];   // 128 x 32 (+pad)

  const int tid  = threadIdx.x;
  const int lane = tid & 31;
  const int wave = tid >> 5;
  const int wm   = wave >> 2;         // 0..1
  const int wn   = wave & 3;          // 0..3
  const int rr   = lane & 15;
  const int half = lane >> 4;

  const int blockN = blockIdx.x * BN;
  const int blockM = blockIdx.y * BM;

  // copy-thread mapping: 16B chunks; A: 256 chunks, W: 512 chunks
  const int arow = tid >> 2, ac = tid & 3;             // A: row 0..63, chunk 0..3

  v8f acc[2][2] = {{{}, {}}, {{}, {}}};

  for (int k0 = 0; k0 < K; k0 += KC) {
    __syncthreads();                      // previous chunk fully consumed
    cp16_g2l(A + (size_t)(blockM + arow) * K + k0 + ac * 8,
             &sA[arow * LDT + ac * 8]);
#pragma unroll
    for (int it = 0; it < 2; ++it) {
      const int idx = tid + it * 256;
      const int row = idx >> 2, c = idx & 3;           // W: row 0..127
      cp16_g2l(W + (size_t)(blockN + row) * K + k0 + c * 8,
               &sW[row * LDT + c * 8]);
    }
    cp_wait();
    __syncthreads();                      // staged tile visible to all waves

    Frag fa[2], fb[2];
#pragma unroll
    for (int tm = 0; tm < 2; ++tm)
      fa[tm] = lds_frag(sA, LDT, wm * 32 + tm * 16 + rr, half);
#pragma unroll
    for (int tn = 0; tn < 2; ++tn)
      fb[tn] = lds_frag(sW, LDT, wn * 32 + tn * 16 + rr, half);

#pragma unroll
    for (int tm = 0; tm < 2; ++tm)
#pragma unroll
      for (int tn = 0; tn < 2; ++tn)
        acc[tm][tn] = __builtin_amdgcn_wmma_f32_16x16x32_bf16(
            false, fa[tm].h, false, fb[tn].h, (short)0, acc[tm][tn],
            false, false);
  }

#pragma unroll
  for (int tm = 0; tm < 2; ++tm) {
#pragma unroll
    for (int tn = 0; tn < 2; ++tn) {
      float* crow = C + (size_t)(blockM + wm * 32 + tm * 16 + half * 8) * N
                      + (blockN + wn * 32 + tn * 16 + rr);
#pragma unroll
      for (int r = 0; r < 8; ++r) crow[(size_t)r * N] = acc[tm][tn][r];
    }
  }
}

// ---------------- simple per-wave WMMA GEMM (for skinny N, e.g. x_proj N=96)
__global__ void k_gemm_simple(const uint16_t* __restrict__ A,
                              const uint16_t* __restrict__ W,
                              float* __restrict__ C,
                              int M, int N, int K) {
  const int nTile = blockIdx.x * 16;
  const int mTile = blockIdx.y * 16;
  const int lane  = threadIdx.x & 31;
  const int rr    = lane & 15;
  const int half  = lane >> 4;

  const uint16_t* arow = A + (size_t)(mTile + rr) * K;
  const uint16_t* wrow = W + (size_t)(nTile + rr) * K;

  v8f acc = {};
  for (int k0 = 0; k0 < K; k0 += 32) {
    Frag ua, ub;
    ua.q[0] = *(const uint4*)(arow + k0 + half * 8);
    ua.q[1] = *(const uint4*)(arow + k0 + 16 + half * 8);
    ub.q[0] = *(const uint4*)(wrow + k0 + half * 8);
    ub.q[1] = *(const uint4*)(wrow + k0 + 16 + half * 8);
    acc = __builtin_amdgcn_wmma_f32_16x16x32_bf16(
        false, ua.h, false, ub.h, (short)0, acc, false, false);
  }

  float* crow = C + (size_t)(mTile + half * 8) * N + (nTile + rr);
#pragma unroll
  for (int r = 0; r < 8; ++r) crow[(size_t)r * N] = acc[r];
}

// ---------------- selective scan ----------------
// grid = B_SZ * (D_INNER/256), block = 256.  Each thread owns channel d of
// batch b; 16 fp32 states in registers.  B_t/C_t staged into LDS 256 steps
// at a time -> inner recurrence is barrier-free.
#define TCHUNK 256
__global__ void __launch_bounds__(256)
k_scan(const float* __restrict__ delta, const float* __restrict__ u,
       const float* __restrict__ xdbl, const float* __restrict__ A_log,
       float* __restrict__ ys) {
  __shared__ float sBC[TCHUNK * 32];      // [t][0..15]=B_t, [t][16..31]=C_t
  const int blocksPerBatch = D_INNER / 256;
  const int b = blockIdx.x / blocksPerBatch;
  const int d = (blockIdx.x % blocksPerBatch) * 256 + threadIdx.x;
  const int tid = threadIdx.x;
  const int XW = DT_RANK + 2 * D_STATE;   // 96

  float a[D_STATE], h[D_STATE];
#pragma unroll
  for (int s = 0; s < D_STATE; ++s) {
    a[s] = -__expf(A_log[(size_t)d * D_STATE + s]);
    h[s] = 0.0f;
  }

  for (int t0 = 0; t0 < L_SEQ; t0 += TCHUNK) {
    __syncthreads();
    for (int idx = tid; idx < TCHUNK * 32; idx += 256) {
      const int tt = idx >> 5, c = idx & 31;
      sBC[idx] = xdbl[(size_t)(b * L_SEQ + t0 + tt) * XW + DT_RANK + c];
    }
    __syncthreads();

    for (int tt = 0; tt < TCHUNK; ++tt) {
      const size_t row = (size_t)(b * L_SEQ + t0 + tt);
      const float dv = delta[row * D_INNER + d];
      const float du = dv * u[row * D_INNER + d];
      const float* bc = &sBC[tt * 32];
      float y = 0.0f;
#pragma unroll
      for (int s = 0; s < D_STATE; ++s) {
        const float dA = __expf(dv * a[s]);
        h[s] = dA * h[s] + du * bc[s];
        y += h[s] * bc[16 + s];
      }
      ys[row * D_INNER + d] = y;
    }
  }
}

// ---------------- launch ----------------

static inline size_t alignup(size_t v) { return (v + 255) & ~(size_t)255; }

extern "C" void kernel_launch(void* const* d_in, const int* in_sizes, int n_in,
                              void* d_out, int out_size, void* d_ws, size_t ws_size,
                              hipStream_t stream) {
  (void)in_sizes; (void)n_in; (void)out_size; (void)ws_size;

  const float* x        = (const float*)d_in[0];   // (B,L,DIM)
  const float* norm_w   = (const float*)d_in[1];   // (DIM)
  const float* in_w     = (const float*)d_in[2];   // (2*D_INNER, DIM)
  const float* conv_w   = (const float*)d_in[3];   // (D_INNER,1,D_CONV)
  const float* conv_b   = (const float*)d_in[4];   // (D_INNER)
  const float* xp_w     = (const float*)d_in[5];   // (96, D_INNER)
  const float* dt_w     = (const float*)d_in[6];   // (D_INNER, DT_RANK)
  const float* dt_b     = (const float*)d_in[7];   // (D_INNER)
  const float* A_log    = (const float*)d_in[8];   // (D_INNER, D_STATE)
  const float* Dp       = (const float*)d_in[9];   // (D_INNER)
  const float* out_w    = (const float*)d_in[10];  // (DIM, D_INNER)
  float* out            = (float*)d_out;           // (B,L,DIM)

  const int XW = DT_RANK + 2 * D_STATE;            // 96

  char* w = (char*)d_ws;
  size_t off = 0;
  auto take = [&](size_t bytes) { char* p = w + off; off = alignup(off + bytes); return p; };

  uint16_t* xn_bf   = (uint16_t*)take((size_t)NTOK * DIM * 2);
  uint16_t* winbf   = (uint16_t*)take((size_t)2 * D_INNER * DIM * 2);
  uint16_t* wxpbf   = (uint16_t*)take((size_t)XW * D_INNER * 2);
  uint16_t* wdtbf   = (uint16_t*)take((size_t)D_INNER * DT_RANK * 2);
  uint16_t* woutbf  = (uint16_t*)take((size_t)DIM * D_INNER * 2);
  float*    xz      = (float*)take((size_t)NTOK * 2 * D_INNER * 4);
  float*    u       = (float*)take((size_t)NTOK * D_INNER * 4);
  uint16_t* u_bf    = (uint16_t*)take((size_t)NTOK * D_INNER * 2);
  float*    xdbl    = (float*)take((size_t)NTOK * XW * 4);
  uint16_t* dt_bf   = (uint16_t*)take((size_t)NTOK * DT_RANK * 2);
  float*    delta   = (float*)take((size_t)NTOK * D_INNER * 4);
  float*    ys      = (float*)take((size_t)NTOK * D_INNER * 4);
  uint16_t* y_bf    = (uint16_t*)take((size_t)NTOK * D_INNER * 2);
  float*    proj    = (float*)take((size_t)NTOK * DIM * 4);

  const int T = 256;
  auto blks = [](int n, int t) { return (n + t - 1) / t; };

  // weights -> bf16
  k_cvt_bf16<<<blks(2 * D_INNER * DIM, T), T, 0, stream>>>(in_w,  winbf,  2 * D_INNER * DIM);
  k_cvt_bf16<<<blks(XW * D_INNER,      T), T, 0, stream>>>(xp_w,  wxpbf,  XW * D_INNER);
  k_cvt_bf16<<<blks(D_INNER * DT_RANK, T), T, 0, stream>>>(dt_w,  wdtbf,  D_INNER * DT_RANK);
  k_cvt_bf16<<<blks(DIM * D_INNER,     T), T, 0, stream>>>(out_w, woutbf, DIM * D_INNER);

  // 1) RMSNorm -> bf16
  k_rmsnorm<<<NTOK, 256, 0, stream>>>(x, norm_w, xn_bf);

  // 2) in_proj: (2048 x 1024) * (4096 x 1024)^T -> xz (2048 x 4096)
  k_gemm_tiled<<<dim3((2 * D_INNER) / BN, NTOK / BM), 256, 0, stream>>>(
      xn_bf, winbf, xz, NTOK, 2 * D_INNER, DIM);

  // 3) depthwise causal conv + SiLU -> u (f32 + bf16)
  k_conv_silu<<<blks(NTOK * D_INNER, T), T, 0, stream>>>(xz, conv_w, conv_b, u, u_bf);

  // 4) x_proj: (2048 x 2048) * (96 x 2048)^T -> x_dbl (2048 x 96)
  k_gemm_simple<<<dim3(XW / 16, NTOK / 16), 32, 0, stream>>>(
      u_bf, wxpbf, xdbl, NTOK, XW, D_INNER);

  // 5) dt slice -> bf16, dt_proj: (2048 x 64) * (2048 x 64)^T -> delta_raw
  k_dt_slice<<<blks(NTOK * DT_RANK, T), T, 0, stream>>>(xdbl, dt_bf);
  k_gemm_tiled<<<dim3(D_INNER / BN, NTOK / BM), 256, 0, stream>>>(
      dt_bf, wdtbf, delta, NTOK, D_INNER, DT_RANK);
  k_softplus_bias<<<blks(NTOK * D_INNER, T), T, 0, stream>>>(delta, dt_b);

  // 6) selective scan
  k_scan<<<B_SZ * (D_INNER / 256), 256, 0, stream>>>(delta, u, xdbl, A_log, ys);

  // 7) gate -> bf16
  k_gate<<<blks(NTOK * D_INNER, T), T, 0, stream>>>(ys, u, Dp, xz, y_bf);

  // 8) out_proj: (2048 x 2048) * (1024 x 2048)^T -> proj, then residual
  k_gemm_tiled<<<dim3(DIM / BN, NTOK / BM), 256, 0, stream>>>(
      y_bf, woutbf, proj, NTOK, DIM, D_INNER);
  k_residual<<<blks(NTOK * DIM, T), T, 0, stream>>>(x, proj, out);
}
